// InformationInteractive_11957188952715
// MI455X (gfx1250) — matrix operand order repaired and checked by reading, not compiled, not run on hardware
//
#include <hip/hip_runtime.h>
#include <cstdint>
#include <cstddef>

// ---------------------------------------------------------------------------
// Types for CDNA5 WMMA (wave32): bf16 A/B fragments (16 elems = 8 VGPRs),
// f32 C/D accumulator (8 VGPRs).
// ---------------------------------------------------------------------------
typedef __attribute__((ext_vector_type(16))) __bf16 v16bf;
typedef __attribute__((ext_vector_type(8)))  __bf16 v8bf;
typedef __attribute__((ext_vector_type(8)))  float  v8f;

constexpr int  kB   = 2;
constexpr int  kN   = 2048;
constexpr int  kC   = 256;
constexpr int  kNbr = 16;     // K nearest neighbors
constexpr int  kH   = 4;      // heads
constexpr int  kD   = 64;     // head dim
constexpr int  kBN  = kB * kN;            // 4096 points total
constexpr long kME  = (long)kBN * kNbr;   // 65536 edges total
constexpr float kEps = 1e-5f;

union V16U { v16bf v; v8bf h[2]; };

// A-fragment (16x32 bf16, MxK): lane (r=l&15, hh=l>>4) holds
// K = hh*8 + [0..7] in VGPR0-3 and K = 16 + hh*8 + [0..7] in VGPR4-7.
// `p` already includes + hh*8.
__device__ __forceinline__ v16bf ld_a(const __bf16* p) {
  V16U u;
  u.h[0] = *reinterpret_cast<const v8bf*>(p);
  u.h[1] = *reinterpret_cast<const v8bf*>(p + 16);
  return u.v;
}
// B-fragment (32x16 bf16, KxN): lane (col=l&15, hh=l>>4) holds
// K = hh*8 + [0..7] in VGPR0-3 and K = 16 + hh*8 + [0..7] in VGPR4-7. `p` already includes + hh*8.
__device__ __forceinline__ v16bf ld_b(const __bf16* p) {
  V16U u;
  u.h[0] = *reinterpret_cast<const v8bf*>(p);
  u.h[1] = *reinterpret_cast<const v8bf*>(p + 16);
  return u.v;
}

// ---------------------------------------------------------------------------
// General bf16 GEMM, f32 accumulate:  Y[Mtot][Nout] = A[Mtot][K] * Bm^T + bias
// Bm stored as [Nout][K] (row per output column). Segmented B for batched
// attention GEMMs: rows are grouped in segments of segRows rows, each segment
// uses its own Bm block of Nout*K elements.
// One wave computes a 32(row) x 64(col) tile (B fragments reused across two
// row sub-tiles -> 8 WMMA per 12 b128 loads); 8 waves / 256-thread block.
// Mtot must be a multiple of 32.
// ---------------------------------------------------------------------------
__global__ __launch_bounds__(256) void gemm_bf16_kernel(
    const __bf16* __restrict__ A, const __bf16* __restrict__ Bm,
    const float* __restrict__ bias, float* __restrict__ Y,
    int Mtot, int Kd, int Nout, int segRows)
{
  const int lane = threadIdx.x & 31;
  const int colTiles = Nout >> 6;
  const long waveId = (long)blockIdx.x * 8 + (threadIdx.x >> 5);
  const long rowTile = waveId / colTiles;
  const int  colTile = (int)(waveId % colTiles);
  const long row0 = rowTile * 32;
  if (row0 >= Mtot) return;
  const int col0 = colTile * 64;
  const int r  = lane & 15;
  const int hh = lane >> 4;

  const __bf16* aP0  = A + (size_t)(row0 + r) * Kd + hh * 8;
  const __bf16* aP1  = aP0 + (size_t)16 * Kd;
  const __bf16* bSeg = Bm + (size_t)(row0 / segRows) * ((size_t)Nout * Kd);
  const __bf16* bP   = bSeg + (size_t)(col0 + r) * Kd + hh * 8;
  const size_t bStride = (size_t)16 * Kd;

  v8f c00 = {}, c01 = {}, c02 = {}, c03 = {};
  v8f c10 = {}, c11 = {}, c12 = {}, c13 = {};
  for (int k0 = 0; k0 < Kd; k0 += 32) {
    v16bf a0 = ld_a(aP0 + k0);
    v16bf a1 = ld_a(aP1 + k0);
    v16bf b0 = ld_b(bP + k0);
    v16bf b1 = ld_b(bP + bStride + k0);
    v16bf b2 = ld_b(bP + 2 * bStride + k0);
    v16bf b3 = ld_b(bP + 3 * bStride + k0);
    __builtin_prefetch(aP0 + k0 + 128, 0, 1);   // global_prefetch_b8
    c00 = __builtin_amdgcn_wmma_f32_16x16x32_bf16(false, a0, false, b0, (short)0, c00, false, false);
    c01 = __builtin_amdgcn_wmma_f32_16x16x32_bf16(false, a0, false, b1, (short)0, c01, false, false);
    c02 = __builtin_amdgcn_wmma_f32_16x16x32_bf16(false, a0, false, b2, (short)0, c02, false, false);
    c03 = __builtin_amdgcn_wmma_f32_16x16x32_bf16(false, a0, false, b3, (short)0, c03, false, false);
    c10 = __builtin_amdgcn_wmma_f32_16x16x32_bf16(false, a1, false, b0, (short)0, c10, false, false);
    c11 = __builtin_amdgcn_wmma_f32_16x16x32_bf16(false, a1, false, b1, (short)0, c11, false, false);
    c12 = __builtin_amdgcn_wmma_f32_16x16x32_bf16(false, a1, false, b2, (short)0, c12, false, false);
    c13 = __builtin_amdgcn_wmma_f32_16x16x32_bf16(false, a1, false, b3, (short)0, c13, false, false);
  }

  float bv0 = 0.f, bv1 = 0.f, bv2 = 0.f, bv3 = 0.f;
  if (bias) {
    bv0 = bias[col0 + r];       bv1 = bias[col0 + 16 + r];
    bv2 = bias[col0 + 32 + r];  bv3 = bias[col0 + 48 + r];
  }
  // C/D layout: acc[j] for lane = element (rowBase + j + 8*hh, col0 + r + 16*t)
#pragma unroll
  for (int j = 0; j < 8; ++j) {
    size_t rr = (size_t)row0 + j + 8 * hh;
    float* y = Y + rr * Nout + col0 + r;
    y[0]  = c00[j] + bv0;
    y[16] = c01[j] + bv1;
    y[32] = c02[j] + bv2;
    y[48] = c03[j] + bv3;
    float* y2 = y + (size_t)16 * Nout;
    y2[0]  = c10[j] + bv0;
    y2[16] = c11[j] + bv1;
    y2[32] = c12[j] + bv2;
    y2[48] = c13[j] + bv3;
  }
}

// ---------------------------------------------------------------------------
// kNN (k=16) over 3D coords (B,3,N). One thread per point, coords in LDS.
// ---------------------------------------------------------------------------
__global__ __launch_bounds__(256) void knn_kernel(const float* __restrict__ coords,
                                                  int* __restrict__ idx)
{
  __shared__ float sx[kN], sy[kN], sz[kN];
  const int b = blockIdx.y;
  const float* cb = coords + (size_t)b * 3 * kN;
  for (int i = threadIdx.x; i < kN; i += blockDim.x) {
    sx[i] = cb[i]; sy[i] = cb[kN + i]; sz[i] = cb[2 * kN + i];
  }
  __syncthreads();
  const int n = blockIdx.x * blockDim.x + threadIdx.x;
  const float px = sx[n], py = sy[n], pz = sz[n];
  float bd[kNbr]; int bi[kNbr];
#pragma unroll
  for (int j = 0; j < kNbr; ++j) { bd[j] = 3.4e38f; bi[j] = 0; }
  for (int m = 0; m < kN; ++m) {
    if (m == n) continue;
    float dx = px - sx[m], dy = py - sy[m], dz = pz - sz[m];
    float d = dx * dx + dy * dy + dz * dz;
    if (d < bd[kNbr - 1]) {
      float cd = d; int ci = m;
#pragma unroll
      for (int j = 0; j < kNbr; ++j) {
        bool sw = cd < bd[j];
        float td = bd[j]; int ti = bi[j];
        bd[j] = sw ? cd : td; bi[j] = sw ? ci : ti;
        cd = sw ? td : cd;   ci = sw ? ti : ci;
      }
    }
  }
  int* o = idx + ((size_t)b * kN + n) * kNbr;
#pragma unroll
  for (int j = 0; j < kNbr; ++j) o[j] = bi[j];
}

// Edge features: G[edge][0:Cin]=center, G[edge][Cin:2Cin]=neigh-center (bf16)
__global__ __launch_bounds__(256) void gather_edges(const float* __restrict__ F,
                                                    const int* __restrict__ idx,
                                                    __bf16* __restrict__ G, int Cin)
{
  const int bn = blockIdx.x;
  const int b  = bn / kN;
  const float* ctr = F + (size_t)bn * Cin;
  const int* id = idx + (size_t)bn * kNbr;
  for (int kk = 0; kk < kNbr; ++kk) {
    const float* ngh = F + ((size_t)b * kN + id[kk]) * Cin;
    __bf16* g = G + ((size_t)bn * kNbr + kk) * (2 * Cin);
    for (int c = threadIdx.x; c < Cin; c += blockDim.x) {
      float cv = ctr[c];
      g[c]       = (__bf16)cv;
      g[Cin + c] = (__bf16)(ngh[c] - cv);
    }
  }
}

// Generic f32 -> bf16 cast / strided concat copy.
__global__ void copy_cast_bf16(const float* __restrict__ src, __bf16* __restrict__ dst,
                               long n, int srcC, int dstC, int dstOff)
{
  long i = (long)blockIdx.x * blockDim.x + threadIdx.x;
  if (i >= n) return;
  long r = i / srcC; int c = (int)(i % srcC);
  dst[r * (long)dstC + dstOff + c] = (__bf16)src[i];
}

// Per-(batch,channel) mean / rstd over rowsPerBatch rows of Y[B*rows][O].
__global__ void col_stats(const float* __restrict__ Y, float2* __restrict__ stats,
                          int rowsPerBatch, int O)
{
  int g = blockIdx.x * blockDim.x + threadIdx.x;
  if (g >= kB * O) return;
  int b = g / O, o = g % O;
  const float* y = Y + (size_t)b * rowsPerBatch * O + o;
  float s = 0.f, s2 = 0.f;
  for (int r = 0; r < rowsPerBatch; ++r) {
    float v = y[(size_t)r * O];
    s += v; s2 += v * v;
  }
  float inv = 1.f / (float)rowsPerBatch;
  float mu = s * inv;
  float var = fmaxf(s2 * inv - mu * mu, 0.f);
  stats[g] = make_float2(mu, rsqrtf(var + kEps));
}

// inorm + leaky-relu + max over k neighbors: Y[B*N*k][O] -> F[B*N][O]
__global__ void norm_lrelu_max(const float* __restrict__ Y, const float2* __restrict__ stats,
                               float* __restrict__ Fout, int O)
{
  long g = (long)blockIdx.x * blockDim.x + threadIdx.x;
  long total = (long)kBN * O;
  if (g >= total) return;
  int o = (int)(g % O);
  long bn = g / O;
  int b = (int)(bn / kN);
  float2 s = stats[b * O + o];
  const float* y = Y + (size_t)bn * kNbr * O + o;
  float m = -3.4e38f;
#pragma unroll
  for (int kk = 0; kk < kNbr; ++kk) {
    float v = (y[(size_t)kk * O] - s.x) * s.y;
    v = v >= 0.f ? v : 0.2f * v;
    m = fmaxf(m, v);
  }
  Fout[g] = m;
}

// inorm + activation, f32 and/or bf16 output. mode 0: leaky-relu, 1: relu.
__global__ void norm_act(const float* __restrict__ Y, const float2* __restrict__ stats,
                         float* __restrict__ outF, __bf16* __restrict__ outB,
                         int O, int mode)
{
  long g = (long)blockIdx.x * blockDim.x + threadIdx.x;
  long total = (long)kBN * O;
  if (g >= total) return;
  int o = (int)(g % O);
  int b = (int)((g / O) / kN);
  float2 s = stats[b * O + o];
  float v = (Y[g] - s.x) * s.y;
  v = (mode == 0) ? (v >= 0.f ? v : 0.2f * v) : fmaxf(v, 0.f);
  if (outF) outF[g] = v;
  if (outB) outB[g] = (__bf16)v;
}

// Row softmax with scale, f32 in -> bf16 probs out. cols <= 2048.
__global__ __launch_bounds__(256) void softmax_rows(const float* __restrict__ S,
                                                    __bf16* __restrict__ P,
                                                    int cols, float scale)
{
  __shared__ float red[256];
  const long row = blockIdx.x;
  const float* s = S + (size_t)row * cols;
  float v[8];
  int cnt = 0;
  float mx = -3.4e38f;
  for (int c = threadIdx.x; c < cols; c += 256) {
    float x = s[c] * scale;
    v[cnt++] = x;
    mx = fmaxf(mx, x);
  }
  red[threadIdx.x] = mx; __syncthreads();
  for (int st = 128; st > 0; st >>= 1) {
    if (threadIdx.x < st) red[threadIdx.x] = fmaxf(red[threadIdx.x], red[threadIdx.x + st]);
    __syncthreads();
  }
  mx = red[0]; __syncthreads();
  float sum = 0.f;
  cnt = 0;
  for (int c = threadIdx.x; c < cols; c += 256) {
    float e = __expf(v[cnt] - mx);
    v[cnt++] = e;
    sum += e;
  }
  red[threadIdx.x] = sum; __syncthreads();
  for (int st = 128; st > 0; st >>= 1) {
    if (threadIdx.x < st) red[threadIdx.x] += red[threadIdx.x + st];
    __syncthreads();
  }
  float inv = 1.f / red[0];
  cnt = 0;
  for (int c = threadIdx.x; c < cols; c += 256)
    P[(size_t)row * cols + c] = (__bf16)(v[cnt++] * inv);
}

// Split Y[B*N][C] (c = d*H + h) into per-head bf16: [B*H][N][D] or transposed [B*H][D][N].
__global__ void split_heads(const float* __restrict__ Y, __bf16* __restrict__ out, int transposed)
{
  long g = (long)blockIdx.x * blockDim.x + threadIdx.x;
  constexpr long total = (long)kB * kH * kN * kD;
  if (g >= total) return;
  int d = (int)(g % kD);
  long t = g / kD;
  int n = (int)(t % kN);
  long t2 = t / kN;
  int h = (int)(t2 % kH);
  int b = (int)(t2 / kH);
  float v = Y[((size_t)b * kN + n) * kC + d * kH + h];
  if (transposed)
    out[(((size_t)(b * kH + h)) * kD + d) * kN + n] = (__bf16)v;
  else
    out[g] = (__bf16)v;
}

// Merge per-head output Yv[B*H][N][D] -> bf16 [B*N][C] with c = d*H + h.
__global__ void merge_heads(const float* __restrict__ Yv, __bf16* __restrict__ out)
{
  long g = (long)blockIdx.x * blockDim.x + threadIdx.x;
  constexpr long total = (long)kBN * kC;
  if (g >= total) return;
  int c = (int)(g % kC);
  long bn = g / kC;
  int n = (int)(bn % kN);
  int b = (int)(bn / kN);
  int h = c % kH, d = c / kH;
  out[g] = (__bf16)Yv[(((size_t)(b * kH + h)) * kN + n) * kD + d];
}

// (B,C,N) <-> point-major (B,N,C)
__global__ void cn_to_nc(const float* __restrict__ X, float* __restrict__ P, int C)
{
  long g = (long)blockIdx.x * blockDim.x + threadIdx.x;
  long total = (long)kB * C * kN;
  if (g >= total) return;
  int n = (int)(g % kN);
  int c = (int)((g / kN) % C);
  int b = (int)(g / ((long)kN * C));
  P[((size_t)b * kN + n) * C + c] = X[g];
}
__global__ void nc_to_cn(const float* __restrict__ P, float* __restrict__ X, int C)
{
  long g = (long)blockIdx.x * blockDim.x + threadIdx.x;
  long total = (long)kB * C * kN;
  if (g >= total) return;
  int n = (int)(g % kN);
  int c = (int)((g / kN) % C);
  int b = (int)(g / ((long)kN * C));
  X[g] = P[((size_t)b * kN + n) * C + c];
}

// ---------------------------------------------------------------------------
// Host orchestration
// ---------------------------------------------------------------------------
extern "C" void kernel_launch(void* const* d_in, const int* in_sizes, int n_in,
                              void* d_out, int out_size, void* d_ws, size_t ws_size,
                              hipStream_t stream)
{
  (void)in_sizes; (void)n_in; (void)out_size; (void)ws_size;
  const float* coords1 = (const float*)d_in[0];
  const float* featsA  = (const float*)d_in[1];
  const float* coords2 = (const float*)d_in[2];
  const float* featsB  = (const float*)d_in[3];

  // Workspace carve-out
  char* ws = (char*)d_ws;
  size_t off = 0;
  auto alloc = [&](size_t bytes) -> void* {
    off = (off + 255) & ~(size_t)255;
    void* p = ws + off;
    off += bytes;
    return p;
  };
  float*  f1a   = (float*) alloc((size_t)kBN * kC * 4);
  float*  f1b   = (float*) alloc((size_t)kBN * kC * 4);
  float*  f2a   = (float*) alloc((size_t)kBN * kC * 4);
  float*  f2b   = (float*) alloc((size_t)kBN * kC * 4);
  float*  F1buf = (float*) alloc((size_t)kBN * 256 * 4);
  float*  F2buf = (float*) alloc((size_t)kBN * 512 * 4);
  float2* stats = (float2*)alloc((size_t)kB * 512 * sizeof(float2));
  __bf16* qh    = (__bf16*)alloc((size_t)kB * kH * kN * kD * 2);
  __bf16* kh    = (__bf16*)alloc((size_t)kB * kH * kN * kD * 2);
  __bf16* vt    = (__bf16*)alloc((size_t)kB * kH * kD * kN * 2);
  __bf16* bfA   = (__bf16*)alloc((size_t)kBN * 1024 * 2);
  __bf16* bfB   = (__bf16*)alloc((size_t)kBN * 512 * 2);
  int*    idxb  = (int*)   alloc((size_t)kB * kN * kNbr * 4);
  __bf16* wbf   = (__bf16*)alloc((size_t)2621440 * 2);
  __bf16* Greg  = (__bf16*)alloc((size_t)kME * 512 * 2);      // edge feats / softmax probs
  float*  Ybig  = (float*) alloc((size_t)kME * 512 * 4);      // 128 MB GEMM output staging

  // bf16 weight images (both layers contiguous per tensor)
  struct WSpec { int inIdx; long elems; };
  const WSpec wspec[9] = {
    {4, 2L*256*512}, {5, 2L*512*512}, {6, 2L*256*1024},
    {7, 2L*256*256}, {9, 2L*256*256}, {11, 2L*256*256}, {13, 2L*256*256},
    {15, 2L*512*512}, {17, 2L*256*512}
  };
  __bf16* wp[9];
  {
    long woff = 0;
    for (int i = 0; i < 9; ++i) {
      wp[i] = wbf + woff;
      long n = wspec[i].elems;
      copy_cast_bf16<<<(int)((n + 255) / 256), 256, 0, stream>>>(
          (const float*)d_in[wspec[i].inIdx], wp[i], n, 1, 1, 0);
      woff += n;
    }
  }
  __bf16 *w1bf = wp[0], *w2bf = wp[1], *w3bf = wp[2];
  __bf16 *wqbf = wp[3], *wkbf = wp[4], *wvbf = wp[5], *wobf = wp[6];
  __bf16 *wm1bf = wp[7], *wm2bf = wp[8];
  const float* bq  = (const float*)d_in[8];
  const float* bk  = (const float*)d_in[10];
  const float* bv  = (const float*)d_in[12];
  const float* bo  = (const float*)d_in[14];
  const float* bm1 = (const float*)d_in[16];
  const float* bm2 = (const float*)d_in[18];

  auto gemm = [&](const __bf16* A, const __bf16* Bm, const float* bias, float* Y,
                  int Mtot, int Kd, int Nout, int segRows) {
    int colTiles = Nout >> 6;
    long waves = (long)(Mtot >> 5) * colTiles;   // 32-row tiles per wave
    int blocks = (int)((waves + 7) >> 3);
    gemm_bf16_kernel<<<blocks, 256, 0, stream>>>(A, Bm, bias, Y, Mtot, Kd, Nout, segRows);
  };
  auto cast = [&](const float* src, __bf16* dst, long n, int srcC, int dstC, int dstOff) {
    copy_cast_bf16<<<(int)((n + 255) / 256), 256, 0, stream>>>(src, dst, n, srcC, dstC, dstOff);
  };

  // ---- one GCN application on one cloud ----
  auto run_gcn = [&](const float* coords, float*& fcur, float*& falt, int li) {
    knn_kernel<<<dim3(kN / 256, kB), 256, 0, stream>>>(coords, idxb);
    // edge conv 1: (65536 x 512) @ W1[256][512]
    gather_edges<<<kBN, 256, 0, stream>>>(fcur, idxb, Greg, kC);
    gemm(Greg, w1bf + (long)li * 256 * 512, nullptr, Ybig, (int)kME, 512, 256, (int)kME);
    col_stats<<<(kB * 256 + 255) / 256, 256, 0, stream>>>(Ybig, stats, kN * kNbr, 256);
    norm_lrelu_max<<<(int)(((long)kBN * 256 + 255) / 256), 256, 0, stream>>>(Ybig, stats, F1buf, 256);
    // edge conv 2: (65536 x 512) @ W2[512][512]
    gather_edges<<<kBN, 256, 0, stream>>>(F1buf, idxb, Greg, 256);
    gemm(Greg, w2bf + (long)li * 512 * 512, nullptr, Ybig, (int)kME, 512, 512, (int)kME);
    col_stats<<<(kB * 512 + 255) / 256, 256, 0, stream>>>(Ybig, stats, kN * kNbr, 512);
    norm_lrelu_max<<<(int)(((long)kBN * 512 + 255) / 256), 256, 0, stream>>>(Ybig, stats, F2buf, 512);
    // concat [feats, f1, f2] -> 1024 ch, @ W3[256][1024]
    cast(fcur,  bfA, (long)kBN * 256, 256, 1024, 0);
    cast(F1buf, bfA, (long)kBN * 256, 256, 1024, 256);
    cast(F2buf, bfA, (long)kBN * 512, 512, 1024, 512);
    gemm(bfA, w3bf + (long)li * 256 * 1024, nullptr, Ybig, kBN, 1024, 256, kBN);
    col_stats<<<(kB * 256 + 255) / 256, 256, 0, stream>>>(Ybig, stats, kN, 256);
    norm_act<<<(int)(((long)kBN * 256 + 255) / 256), 256, 0, stream>>>(
        Ybig, stats, falt, ((__bf16*)nullptr), 256, 0);
    float* t = fcur; fcur = falt; falt = t;
  };

  // ---- one cross-attention application: fq attends to fkv ----
  auto run_attn = [&](float*& fq, float*& fkv, float*& falt, int li) {
    __bf16* fqb  = bfA;
    __bf16* fkb  = bfA + (size_t)kBN * kC;
    __bf16* attb = bfA + (size_t)2 * kBN * kC;
    cast(fq,  fqb, (long)kBN * kC, kC, kC, 0);
    cast(fkv, fkb, (long)kBN * kC, kC, kC, 0);
    const long hd = (long)kB * kH * kN * kD;
    // Q / K / V projections + head split
    gemm(fqb, wqbf + (long)li * kC * kC, bq + (long)li * kC, Ybig, kBN, kC, kC, kBN);
    split_heads<<<(int)((hd + 255) / 256), 256, 0, stream>>>(Ybig, qh, 0);
    gemm(fkb, wkbf + (long)li * kC * kC, bk + (long)li * kC, Ybig, kBN, kC, kC, kBN);
    split_heads<<<(int)((hd + 255) / 256), 256, 0, stream>>>(Ybig, kh, 0);
    gemm(fkb, wvbf + (long)li * kC * kC, bv + (long)li * kC, Ybig, kBN, kC, kC, kBN);
    split_heads<<<(int)((hd + 255) / 256), 256, 0, stream>>>(Ybig, vt, 1);
    // scores = Q Kᵀ (batched per head), softmax(/sqrt(d)), P V
    gemm(qh, kh, nullptr, Ybig, kB * kH * kN, kD, kN, kN);
    softmax_rows<<<kB * kH * kN, 256, 0, stream>>>(Ybig, Greg, kN, 0.125f);
    gemm(Greg, vt, nullptr, Ybig, kB * kH * kN, kN, kD, kN);
    merge_heads<<<(int)(((long)kBN * kC + 255) / 256), 256, 0, stream>>>(Ybig, attb);
    // output projection
    gemm(attb, wobf + (long)li * kC * kC, bo + (long)li * kC, Ybig, kBN, kC, kC, kBN);
    // MLP: concat [fq, att] -> 512 -> Wm1(+inorm+relu) -> Wm2
    cast(fq,   bfB, (long)kBN * kC, kC, 512, 0);
    cast(Ybig, bfB, (long)kBN * kC, kC, 512, kC);
    gemm(bfB, wm1bf + (long)li * 512 * 512, bm1 + (long)li * 512, Ybig, kBN, 512, 512, kBN);
    col_stats<<<(kB * 512 + 255) / 256, 256, 0, stream>>>(Ybig, stats, kN, 512);
    norm_act<<<(int)(((long)kBN * 512 + 255) / 256), 256, 0, stream>>>(
        Ybig, stats, (float*)nullptr, bfA, 512, 1);
    gemm(bfA, wm2bf + (long)li * kC * 512, bm2 + (long)li * kC, falt, kBN, 512, kC, kBN);
    float* t = fq; fq = falt; falt = t;
  };

  // ---- forward pass ----
  const long elemsCN = (long)kB * kC * kN;
  cn_to_nc<<<(int)((elemsCN + 255) / 256), 256, 0, stream>>>(featsA, f1a, kC);
  cn_to_nc<<<(int)((elemsCN + 255) / 256), 256, 0, stream>>>(featsB, f2a, kC);
  float *f1cur = f1a, *f1alt = f1b, *f2cur = f2a, *f2alt = f2b;

  // layer 0: gcn
  run_gcn(coords1, f1cur, f1alt, 0);
  run_gcn(coords2, f2cur, f2alt, 0);
  // layer 1: cross-attn (feats1 first, then feats2 against *updated* feats1)
  run_attn(f1cur, f2cur, f1alt, 0);
  run_attn(f2cur, f1cur, f2alt, 0);
  // layer 2: gcn
  run_gcn(coords1, f1cur, f1alt, 1);
  run_gcn(coords2, f2cur, f2alt, 1);
  // layer 3: cross-attn
  run_attn(f1cur, f2cur, f1alt, 1);
  run_attn(f2cur, f1cur, f2alt, 1);

  // outputs back to (B,C,N), concatenated
  float* out = (float*)d_out;
  nc_to_cn<<<(int)((elemsCN + 255) / 256), 256, 0, stream>>>(f1cur, out, kC);
  nc_to_cn<<<(int)((elemsCN + 255) / 256), 256, 0, stream>>>(f2cur, out + elemsCN, kC);
}
